// LocalSelfAttention2D_70514773066580
// MI455X (gfx1250) — matrix operand order repaired and checked by reading, not compiled, not run on hardware
//
#include <hip/hip_runtime.h>
#include <hip/hip_bf16.h>
#include <math.h>

typedef __attribute__((ext_vector_type(16))) _Float16 v16h;
typedef __attribute__((ext_vector_type(8)))  _Float16 v8h;
typedef __attribute__((ext_vector_type(8)))  float    v8f;

// Problem constants (reference: B=2, Cin=Cout=256, H=W=64, KS=7, P=3)
#define BATCH   2
#define CH      256
#define HH      64
#define WW      64
#define HW      (HH * WW)          // 4096
#define NPIX    (BATCH * HW)       // 8192
#define PAD     3
#define HP      (HH + 2 * PAD)     // 70
#define WP      (WW + 2 * PAD)     // 70
#define PLANE   (HP * WP)          // 4900
#define K2      49
#define EPSBN   1e-5f

// Workspace byte offsets (all 256B aligned)
#define XT_OFF   ((size_t)0)                       // f16 [NPIX][CH]        4 MiB
#define WH_OFF   ((size_t)4194304)                 // f16 [768][CH]         384 KiB
#define BC_OFF   ((size_t)4587520)                 // f32 [768]             3 KiB
#define Q_OFF    ((size_t)4590592)                 // f32 [B][CH][HW]       8 MiB
#define KP_OFF   ((size_t)12979200)                // f32 [B][CH][HP][WP]   ~9.6 MiB
#define VP_OFF   ((size_t)23014400)                // f32 [B][CH][HP][WP]   ~9.6 MiB
#define OP_OFF   ((size_t)33049600)                // f32 [B][CH][HW]       8 MiB (pre-BN)
#define MV_OFF   ((size_t)41438208)                // f32 [2][CH] mean, inv_std

// ---------------------------------------------------------------------------
// (a) x [B,C,HW] f32  ->  xt [B*HW, C] f16   (transpose so WMMA B-operand
//     is 16 contiguous K-halfs per lane = one 32-byte load)
// ---------------------------------------------------------------------------
__global__ void lsa_pack_x(const float* __restrict__ x, _Float16* __restrict__ xt) {
    int idx = blockIdx.x * blockDim.x + threadIdx.x;          // over B*CH*HW
    if (idx >= BATCH * CH * HW) return;
    int hw = idx & (HW - 1);
    int c  = (idx >> 12) & (CH - 1);
    int b  = idx >> 20;
    xt[((size_t)(b * HW + hw)) * CH + c] = (_Float16)x[idx];
}

// ---------------------------------------------------------------------------
// (b) concat wq|wk|wv -> f16 [768][256], biases -> f32 [768]
// ---------------------------------------------------------------------------
__global__ void lsa_pack_w(const float* __restrict__ wq, const float* __restrict__ wk,
                           const float* __restrict__ wv, const float* __restrict__ bq,
                           const float* __restrict__ bk, const float* __restrict__ bv,
                           _Float16* __restrict__ wh, float* __restrict__ bc) {
    int idx = blockIdx.x * blockDim.x + threadIdx.x;          // over 768*256
    if (idx >= 3 * CH * CH) return;
    int o = idx >> 8;                                         // 0..767
    int c = idx & 255;
    float v;
    if      (o < CH)     v = wq[o * CH + c];
    else if (o < 2 * CH) v = wk[(o - CH) * CH + c];
    else                 v = wv[(o - 2 * CH) * CH + c];
    wh[idx] = (_Float16)v;
    if (idx < 3 * CH) {
        float bb;
        if      (idx < CH)     bb = bq[idx];
        else if (idx < 2 * CH) bb = bk[idx - CH];
        else                   bb = bv[idx - 2 * CH];
        bc[idx] = bb;
    }
}

// ---------------------------------------------------------------------------
// (d) Fused QKV projection GEMM via WMMA, register-blocked along N.
//     One wave32 computes a 16(out-ch) x 64(pixel) strip: 4 accumulators,
//     A loaded ONCE per K-step and reused by 4 WMMAs (4 wmma : 5 loads).
//     K=256 in 8 steps of 32  ->  32 v_wmma_f32_16x16x32_f16 per wave.
//     A layout: lane L holds row M=L&15;
//       lanes 0-15 hold K 0..15, lanes 16-31 hold K 16..31 (contiguous halfs)
//     B layout (16-bit B 32x16): lane L holds col N=L&15;
//       lanes 0-15 hold K 0..15, lanes 16-31 hold K 16..31 (contiguous halfs)
//     C/D layout: acc[i] on lane L = row (i + 8*(L>=16)), col (L&15).
//     q rows go to q[], k rows to padded kp[], v rows to padded vp[].
// ---------------------------------------------------------------------------
__global__ void lsa_qkv_wmma(const _Float16* __restrict__ xt,
                             const _Float16* __restrict__ wh,
                             const float* __restrict__ bc,
                             float* __restrict__ q,
                             float* __restrict__ kp,
                             float* __restrict__ vp) {
    const int lane  = threadIdx.x;            // 0..31
    const int n0    = blockIdx.x << 6;        // pixel strip base (128 strips)
    const int m0    = blockIdx.y << 4;        // out-channel tile base (48 tiles)
    const int mrow  = lane & 15;
    const int kbA   = (lane >> 4) << 4;       // 0 or 16
    const int kbB   = (lane >> 4) << 4;       // 0 or 16

    const _Float16* wrow = wh + (size_t)(m0 + mrow) * CH + kbA;
    const _Float16* xcol = xt + (size_t)(n0 + mrow) * CH + kbB;

    v8f acc[4] = {};
#pragma unroll
    for (int kk = 0; kk < CH; kk += 32) {
        if (kk + 32 < CH) {
            __builtin_prefetch(xcol + kk + 32, 0, 0);   // global_prefetch_b8
        }
        v16h a;                                           // shared across 4 WMMAs
        a = *(const v16h*)(wrow + kk);                    // K = kbA .. kbA+15
        v16h bm[4];
#pragma unroll
        for (int s = 0; s < 4; ++s)
            bm[s] = *(const v16h*)(xcol + (size_t)(s * 16) * CH + kk);
#pragma unroll
        for (int s = 0; s < 4; ++s)
            acc[s] = __builtin_amdgcn_wmma_f32_16x16x32_f16(
                         false, a, false, bm[s], (short)0, acc[s], false, false);
    }

    const int mh = (lane >> 4) << 3;          // row-half offset 0 or 8
#pragma unroll
    for (int s = 0; s < 4; ++s) {
        const int pix = n0 + s * 16 + mrow;   // column of this lane in sub-tile s
        const int bb  = pix >> 12;
        const int hw  = pix & (HW - 1);
        const int ypad = ((hw >> 6) + PAD) * WP + ((hw & 63) + PAD);
#pragma unroll
        for (int i = 0; i < 8; ++i) {
            const int m = m0 + mh + i;
            const float val = acc[s][i] + bc[m];
            if (m < CH) {
                q[((size_t)(bb * CH + m)) * HW + hw] = val;
            } else if (m < 2 * CH) {
                kp[((size_t)(bb * CH + (m - CH))) * PLANE + ypad] = val;
            } else {
                vp[((size_t)(bb * CH + (m - 2 * CH))) * PLANE + ypad] = val;
            }
        }
    }
}

// ---------------------------------------------------------------------------
// (e) Per-pixel scrambled-window attention (faithful to the raw reshape):
//     flat = p*4096 + h*64 + w ; h0=flat/3136 ; w0=(flat%3136)/49 ;
//     (i,j)=divmod(flat%49,7) ; sample padded planes at (h0+i, w0+j).
//     Block = 64 threads per pixel. q cached in LDS; 49 scores -> masked
//     softmax -> weighted sum over v for all 256 channels.
// ---------------------------------------------------------------------------
__global__ void lsa_attn(const float* __restrict__ q,
                         const float* __restrict__ kp,
                         const float* __restrict__ vp,
                         float* __restrict__ outp) {
    __shared__ float qs[CH];
    __shared__ int   offs[K2];
    __shared__ float fr[K2];
    __shared__ float at[K2];

    const int pix = blockIdx.x;               // 0..8191
    const int t   = threadIdx.x;              // 0..63
    const int bb  = pix >> 12;
    const int hw  = pix & (HW - 1);
    const size_t qbase = (size_t)bb * CH * HW + hw;
    const size_t pbase = (size_t)bb * CH * PLANE;

#pragma unroll
    for (int r = 0; r < CH / 64; ++r) qs[t + 64 * r] = q[qbase + (size_t)(t + 64 * r) * HW];
    __syncthreads();

    if (t < K2) {
        const int flat = t * HW + hw;
        const int h0   = flat / (WW * K2);            // /3136
        int rem        = flat - h0 * (WW * K2);
        const int w0   = rem / K2;
        rem            = rem - w0 * K2;
        const int ii   = rem / 7;
        const int jj   = rem - ii * 7;
        const int off  = (h0 + ii) * WP + (w0 + jj);
        offs[t] = off;
        float s = 0.f;
        const float* kcol = kp + pbase + off;
#pragma unroll 4
        for (int c = 0; c < CH; ++c) s += qs[c] * kcol[(size_t)c * PLANE];
        fr[t] = (s != 0.f) ? s : -__builtin_inff();   // masked_fill(frac==0, -inf)
    }
    __syncthreads();

    // redundant per-thread max/sum over 49 shared values (cheap, divergence-free)
    float mx = -__builtin_inff();
    for (int p = 0; p < K2; ++p) mx = fmaxf(mx, fr[p]);
    float sm = 0.f;
    for (int p = 0; p < K2; ++p) sm += __expf(fr[p] - mx);
    if (t < K2) at[t] = __expf(fr[t] - mx) / sm;
    __syncthreads();

#pragma unroll
    for (int r = 0; r < CH / 64; ++r) {
        const int c = t + 64 * r;
        const float* vcol = vp + pbase + (size_t)c * PLANE;
        float o = 0.f;
#pragma unroll 7
        for (int p = 0; p < K2; ++p) o += at[p] * vcol[offs[p]];
        outp[(size_t)(bb * CH + c) * HW + hw] = o;
    }
}

// ---------------------------------------------------------------------------
// (f1) BN stats: one block per channel, reduce 8192 samples -> mean, inv_std
// ---------------------------------------------------------------------------
__global__ void lsa_bnstats(const float* __restrict__ outp, float* __restrict__ mv) {
    __shared__ float ssum[256];
    __shared__ float ssq[256];
    const int c = blockIdx.x;
    const int t = threadIdx.x;
    float s = 0.f, s2 = 0.f;
    for (int n = t; n < NPIX; n += 256) {
        const int bb = n >> 12, hw = n & (HW - 1);
        const float v = outp[(size_t)(bb * CH + c) * HW + hw];
        s += v; s2 += v * v;
    }
    ssum[t] = s; ssq[t] = s2;
    __syncthreads();
    for (int w = 128; w > 0; w >>= 1) {
        if (t < w) { ssum[t] += ssum[t + w]; ssq[t] += ssq[t + w]; }
        __syncthreads();
    }
    if (t == 0) {
        const float mean = ssum[0] * (1.f / NPIX);
        const float var  = ssq[0] * (1.f / NPIX) - mean * mean;   // biased, as reference
        mv[c]      = mean;
        mv[CH + c] = rsqrtf(var + EPSBN);
    }
}

// ---------------------------------------------------------------------------
// (f2) normalize + affine + ReLU -> d_out [B,C,H,W] f32
// ---------------------------------------------------------------------------
__global__ void lsa_bnapply(const float* __restrict__ outp, const float* __restrict__ mv,
                            const float* __restrict__ gamma, const float* __restrict__ beta,
                            float* __restrict__ out) {
    const int idx = blockIdx.x * blockDim.x + threadIdx.x;
    if (idx >= BATCH * CH * HW) return;
    const int c = (idx >> 12) & (CH - 1);
    const float v = (outp[idx] - mv[c]) * mv[CH + c] * gamma[c] + beta[c];
    out[idx] = v > 0.f ? v : 0.f;
}

// ---------------------------------------------------------------------------
extern "C" void kernel_launch(void* const* d_in, const int* in_sizes, int n_in,
                              void* d_out, int out_size, void* d_ws, size_t ws_size,
                              hipStream_t stream) {
    const float* x     = (const float*)d_in[0];
    const float* wq    = (const float*)d_in[1];
    const float* bq    = (const float*)d_in[2];
    const float* wk    = (const float*)d_in[3];
    const float* bk    = (const float*)d_in[4];
    const float* wv    = (const float*)d_in[5];
    const float* bv    = (const float*)d_in[6];
    const float* gamma = (const float*)d_in[7];
    const float* beta  = (const float*)d_in[8];
    float* out = (float*)d_out;

    char* ws = (char*)d_ws;
    _Float16* xt  = (_Float16*)(ws + XT_OFF);
    _Float16* wh  = (_Float16*)(ws + WH_OFF);
    float*    bc  = (float*)(ws + BC_OFF);
    float*    q   = (float*)(ws + Q_OFF);
    float*    kp  = (float*)(ws + KP_OFF);
    float*    vp  = (float*)(ws + VP_OFF);
    float*    op  = (float*)(ws + OP_OFF);
    float*    mv  = (float*)(ws + MV_OFF);

    // padded planes must be zero outside the interior (graph-capturable memset)
    hipMemsetAsync(kp, 0, (size_t)BATCH * CH * PLANE * sizeof(float), stream);
    hipMemsetAsync(vp, 0, (size_t)BATCH * CH * PLANE * sizeof(float), stream);

    lsa_pack_x<<<(BATCH * CH * HW + 255) / 256, 256, 0, stream>>>(x, xt);
    lsa_pack_w<<<(3 * CH * CH + 255) / 256, 256, 0, stream>>>(wq, wk, wv, bq, bk, bv, wh, bc);

    dim3 ggrid(NPIX / 64, (3 * CH) / 16);     // 128 x 48 strips, one wave each
    lsa_qkv_wmma<<<ggrid, 32, 0, stream>>>(xt, wh, bc, q, kp, vp);

    lsa_attn<<<NPIX, 64, 0, stream>>>(q, kp, vp, op);
    lsa_bnstats<<<CH, 256, 0, stream>>>(op, mv);
    lsa_bnapply<<<(BATCH * CH * HW + 255) / 256, 256, 0, stream>>>(op, mv, gamma, beta, out);
}
